// DS3M_41695542510372
// MI455X (gfx1250) — matrix operand order repaired and checked by reading, not compiled, not run on hardware
//
#include <hip/hip_runtime.h>
#include <hip/hip_bf16.h>
#include <math.h>
#include <stdint.h>

// ---------------------------------------------------------------------------
// DS3M on MI455X (gfx1250, wave32). GEMMs via v_wmma_f32_16x16x32_f16;
// GRU input tiles staged into LDS with the Tensor Data Mover
// (tensor_load_to_lds + s_wait_tensorcnt), double-buffered in the fwd GRU.
// Pipeline (all on `stream`, in-order):
//   K0  prep: Trans (2x2), weight f16 conversion / packing, x -> f16
//   K1  forward GRU  (16 WGs x 32 batch rows, 256 sequential steps)
//   K2  backward GRU (reads h_all, runs t = 255..0, TDM with LDS padding)
//   K3  scan (32 WGs x 16 batch rows): regime posterior + threefry sampling +
//       z/e MLPs (WMMA) + losses
//   K4  deterministic loss finalize
// ---------------------------------------------------------------------------

typedef __attribute__((ext_vector_type(16))) _Float16 v16h;
typedef __attribute__((ext_vector_type(8)))  float    v8f;
typedef __attribute__((ext_vector_type(8)))  _Float16 h8v;
typedef unsigned int u32x4v __attribute__((ext_vector_type(4)));
typedef int          i32x8v __attribute__((ext_vector_type(8)));
typedef int          i32x4v __attribute__((ext_vector_type(4)));

#define T_LEN 256
#define BATCH 512
#define XD    64
#define HD    256
#define ZD    64
#define G3    768           // 3*H gate width
#define KIB   288           // padded K for backward GRU input: [h(256)|y(1)|pad]

union AFrag { v16h v; h8v h[2]; };

// A-fragment (16x32 f16): lane L<16 -> row L, K=k0..k0+15 contiguous;
// lane L>=16 -> row L-16, K=k0+16..k0+31. base points at (row0,k0).
__device__ inline v16h load_a(const _Float16* base, int ld, int lane) {
  const int r  = lane & 15;
  const int ko = (lane >> 4) << 4;
  const _Float16* q = base + (size_t)r * ld + ko;
  AFrag a;
  a.h[0] = *(const h8v*)(q);
  a.h[1] = *(const h8v*)(q + 8);
  return a.v;
}
// B-fragment (32x16 f16) from a [out][in] packed matrix: lane L<16 -> col L,
// K=k0..k0+15 contiguous; lane L>=16 -> col L-16, K=k0+16..k0+31.
__device__ inline v16h load_b(const _Float16* base, int ld, int lane) {
  const int n  = lane & 15;
  const int ko = (lane >> 4) << 4;
  const _Float16* q = base + (size_t)n * ld + ko;
  AFrag b;
  b.h[0] = *(const h8v*)(q);
  b.h[1] = *(const h8v*)(q + 8);
  return b.v;
}
__device__ inline v8f wmma_f16(v16h a, v16h b, v8f c) {
  return __builtin_amdgcn_wmma_f32_16x16x32_f16(false, a, false, b, (short)0, c,
                                                false, false);
}
__device__ inline float sigm(float x)      { return 1.f / (1.f + expf(-x)); }
__device__ inline float softplusf(float x) { return fmaxf(x, 0.f) + log1pf(expf(-fabsf(x))); }

// --- Tensor Data Mover: 2D f16 tile global->LDS (ISA 08_async_tensor §8) ----
// Descriptor group0: count=1 | lds_addr | global_addr[56:0] | type=2.
// Group1: data_size=2B + optional LDS pad config, tensor/tile dims, stride.
// pad_flags: (pad_en<<20)|(pad_interval<<22)|(pad_amount<<25).
// This toolchain's builtin takes 6 args (g0 v4u, g1 v8i, g2 v4i, g3 v4i,
// extra v8i, cpol) -- groups 2/3 and the trailing group are zero for 2D.
__device__ inline void tdm_load_2d(unsigned lds_addr, const void* gptr,
                                   unsigned tile_d0, unsigned tile_d1,
                                   unsigned stride0, unsigned pad_flags) {
  unsigned long long ga = (unsigned long long)(uintptr_t)gptr;
  const unsigned td0 = stride0;        // tensor dim0 = row length (elements)
  const unsigned td1 = 0x00020000u;    // tensor dim1: 131072 rows (ample)
  u32x4v g0;
  g0[0] = 1u;                                           // count=1, user mode
  g0[1] = lds_addr;                                     // LDS byte address
  g0[2] = (unsigned)(ga & 0xFFFFFFFFu);                 // global addr [31:0]
  g0[3] = (unsigned)((ga >> 32) & 0x01FFFFFFu) | (2u << 30); // [56:32]|type=2
  i32x8v g1;
  g1[0] = (int)((1u << 16) | pad_flags);                // data_size=2B
  g1[1] = (int)((td0 & 0xFFFFu) << 16);                 // tensor_dim0 lo
  g1[2] = (int)(((td0 >> 16) & 0xFFFFu) | ((td1 & 0xFFFFu) << 16));
  g1[3] = (int)(((td1 >> 16) & 0xFFFFu) | (tile_d0 << 16));  // tile_dim0
  g1[4] = (int)(tile_d1 & 0xFFFFu);                     // tile_dim1 (dim2=0)
  g1[5] = (int)stride0;                                 // dim0 stride lo32
  g1[6] = 0;
  g1[7] = 0;
  i32x4v gz4 = {0, 0, 0, 0};                            // 2D: groups 2/3 unused
  i32x8v gz8 = {0, 0, 0, 0, 0, 0, 0, 0};
  __builtin_amdgcn_tensor_load_to_lds(g0, g1, gz4, gz4, gz8, 0);
}

// JAX threefry2x32 (20 rounds), matching jax._src.prng exactly.
__device__ inline void threefry2x32(unsigned k0, unsigned k1,
                                    unsigned x0, unsigned x1,
                                    unsigned& o0, unsigned& o1) {
  const unsigned rot[8] = {13u,15u,26u,6u, 17u,29u,16u,24u};
  unsigned ks[3] = {k0, k1, k0 ^ k1 ^ 0x1BD11BDAu};
  unsigned a = x0 + k0, b = x1 + k1;
  #pragma unroll
  for (int g = 0; g < 5; ++g) {
    const unsigned* r = (g & 1) ? (rot + 4) : rot;
    #pragma unroll
    for (int i = 0; i < 4; ++i) {
      a += b;
      b = (b << r[i]) | (b >> (32u - r[i]));
      b ^= a;
    }
    a += ks[(g + 1) % 3];
    b += ks[(g + 2) % 3] + (unsigned)(g + 1);
  }
  o0 = a; o1 = b;
}

// ---------------------------------------------------------------------------
// K0: prep kernels
// ---------------------------------------------------------------------------
__global__ void k_trans(const float* __restrict__ Wdprior,
                        const float* __restrict__ bdprior,
                        float* __restrict__ TransWs, float* __restrict__ out) {
  if (threadIdx.x == 0 && blockIdx.x == 0) {
    const float Ti[2][2] = {{0.95f, 0.05f}, {0.05f, 0.95f}};
    for (int i = 0; i < 2; ++i) {
      float l0 = Ti[i][0]*Wdprior[0] + Ti[i][1]*Wdprior[2] + bdprior[0];
      float l1 = Ti[i][0]*Wdprior[1] + Ti[i][1]*Wdprior[3] + bdprior[1];
      float m = fmaxf(l0, l1);
      float e0 = expf(l0 - m), e1 = expf(l1 - m), s = e0 + e1;
      TransWs[i*2+0] = e0/s*0.5f + (i == 0 ? 0.5f : 0.f);
      TransWs[i*2+1] = e1/s*0.5f + (i == 1 ? 0.5f : 0.f);
    }
    out[0] = 0.f; out[1] = 0.f; out[2] = 0.f;
  }
}
__global__ void k_cvt(const float* __restrict__ src, _Float16* __restrict__ dst, int n) {
  for (int i = blockIdx.x*blockDim.x + threadIdx.x; i < n; i += gridDim.x*blockDim.x)
    dst[i] = (_Float16)src[i];
}
// [2][I][O] f32 -> [2][O][I] f16 (WMMA B-fragment layout)
__global__ void k_packT(const float* __restrict__ src, _Float16* __restrict__ dst,
                        int I, int O) {
  const int n = 2 * I * O;
  for (int idx = blockIdx.x*blockDim.x + threadIdx.x; idx < n; idx += gridDim.x*blockDim.x) {
    int k = idx / (I*O), rem = idx % (I*O), o = rem / I, i = rem % I;
    dst[idx] = (_Float16)src[((size_t)k*I + i)*O + o];
  }
}
// Wih_b [768][257] f32 -> [768][288] f16 with columns reordered to
// [h(256) | y(1) | pad(31)] so the h-part is a TDM-friendly contiguous tile.
__global__ void k_packwib(const float* __restrict__ w, _Float16* __restrict__ dst) {
  const int n = G3 * KIB;
  for (int idx = blockIdx.x*blockDim.x + threadIdx.x; idx < n; idx += gridDim.x*blockDim.x) {
    int o = idx / KIB, c = idx % KIB;
    float v = 0.f;
    if (c < 256)       v = w[(size_t)o*257 + c + 1];   // h columns
    else if (c == 256) v = w[(size_t)o*257];           // y column
    dst[idx] = (_Float16)v;
  }
}

// ---------------------------------------------------------------------------
// K1: forward GRU. 32 batch rows / WG, h state f16 double-buffered in LDS,
// x tiles streamed by the TDM (double-buffered, overlapped with WMMA).
// ---------------------------------------------------------------------------
__global__ __launch_bounds__(256) void gru_fwd(
    const _Float16* __restrict__ x16, const float* __restrict__ bih,
    const float* __restrict__ bhh, const _Float16* __restrict__ Wih16,
    const _Float16* __restrict__ Whh16, _Float16* __restrict__ hall) {
  __shared__ _Float16 hbuf[2][32][HD];
  __shared__ _Float16 xt2[2][32][XD];
  const int b0 = blockIdx.x * 32;
  const int tid = threadIdx.x, lane = tid & 31, wave = tid >> 5;
  const unsigned ldsX0 = (unsigned)(uintptr_t)&xt2[0][0][0];
  const unsigned ldsX1 = (unsigned)(uintptr_t)&xt2[1][0][0];
  for (int i = tid; i < 32*HD; i += 256) ((_Float16*)hbuf)[i] = (_Float16)0.f;
  if (wave == 0)                                  // prime tile t=0 -> buf 0
    tdm_load_2d(ldsX0, x16 + (size_t)b0*XD, XD, 32, XD, 0);
  __syncthreads();
  int cur = 0, buf = 0;
  for (int t = 0; t < T_LEN; ++t) {
    if (wave == 0) __builtin_amdgcn_s_wait_tensorcnt(0);  // tile t landed
    __syncthreads();
    if (wave == 0 && t + 1 < T_LEN)               // overlap TDM for t+1
      tdm_load_2d(buf ? ldsX0 : ldsX1,
                  x16 + ((size_t)(t+1)*BATCH + b0)*XD, XD, 32, XD, 0);
    for (int u = wave; u < 32; u += 8) {          // 2 M-tiles x 16 hid blocks
      const int m0 = (u >> 4) * 16, j0 = (u & 15) * 16;
      v8f aR = {}, aU = {}, aNi = {}, aNh = {};
      for (int kc = 0; kc < 2; ++kc) {            // K over X = 64
        v16h a = load_a(&xt2[buf][m0][kc*32], XD, lane);
        aR  = wmma_f16(a, load_b(Wih16 + (size_t)(      j0)*XD + kc*32, XD, lane), aR);
        aU  = wmma_f16(a, load_b(Wih16 + (size_t)(256 + j0)*XD + kc*32, XD, lane), aU);
        aNi = wmma_f16(a, load_b(Wih16 + (size_t)(512 + j0)*XD + kc*32, XD, lane), aNi);
      }
      for (int kc = 0; kc < 8; ++kc) {            // K over H = 256
        v16h a = load_a(&hbuf[cur][m0][kc*32], HD, lane);
        aR  = wmma_f16(a, load_b(Whh16 + (size_t)(      j0)*HD + kc*32, HD, lane), aR);
        aU  = wmma_f16(a, load_b(Whh16 + (size_t)(256 + j0)*HD + kc*32, HD, lane), aU);
        aNh = wmma_f16(a, load_b(Whh16 + (size_t)(512 + j0)*HD + kc*32, HD, lane), aNh);
      }
      const int j  = j0 + (lane & 15);
      const float bR  = bih[j] + bhh[j];
      const float bU  = bih[256+j] + bhh[256+j];
      const float bNi = bih[512+j], bNh = bhh[512+j];
      const int mb = m0 + ((lane >> 4) << 3);
      #pragma unroll
      for (int i = 0; i < 8; ++i) {               // C layout: M=i+8*hi, N=lane&15
        const int r = mb + i;
        float rr = sigm(aR[i] + bR);
        float uu = sigm(aU[i] + bU);
        float nn = tanhf(aNi[i] + bNi + rr*(aNh[i] + bNh));
        float hnew = (1.f - uu)*nn + uu*(float)hbuf[cur][r][j];
        hbuf[cur^1][r][j] = (_Float16)hnew;
        hall[((size_t)t*BATCH + b0 + r)*HD + j] = (_Float16)hnew;
      }
    }
    __syncthreads();
    cur ^= 1;
    buf ^= 1;
  }
}

// ---------------------------------------------------------------------------
// K2: backward GRU over [h_all(256)|y(1)|pad], t = 255..0. The 32x256 f16
// h-tile is TDM-loaded each step with LDS padding (row pitch 288 halves:
// pad_interval=128 DWORDs -> code 6, pad_amount=16 DWORDs -> code 15).
// ---------------------------------------------------------------------------
__global__ __launch_bounds__(256) void gru_bwd(
    const float* __restrict__ y, const float* __restrict__ bih,
    const float* __restrict__ bhh, const _Float16* __restrict__ Wib16,
    const _Float16* __restrict__ Whb16, const _Float16* __restrict__ hall,
    _Float16* __restrict__ Aall) {
  __shared__ _Float16 hbuf[2][32][HD];
  __shared__ _Float16 xt[32][KIB];
  const int b0 = blockIdx.x * 32;
  const int tid = threadIdx.x, lane = tid & 31, wave = tid >> 5;
  const unsigned ldsXT = (unsigned)(uintptr_t)&xt[0][0];
  const unsigned padflags = (1u << 20) | (6u << 22) | (15u << 25);
  for (int i = tid; i < 32*HD; i += 256) ((_Float16*)hbuf)[i] = (_Float16)0.f;
  for (int i = tid; i < 32*32; i += 256) {        // pad columns stay zero
    int r = i >> 5, c = 256 + (i & 31);
    xt[r][c] = (_Float16)0.f;
  }
  __syncthreads();
  int cur = 0;
  for (int tt = 0; tt < T_LEN; ++tt) {
    const int t = T_LEN - 1 - tt;
    if (wave == 0) {
      tdm_load_2d(ldsXT, hall + ((size_t)t*BATCH + b0)*HD, HD, 32, HD, padflags);
      __builtin_amdgcn_s_wait_tensorcnt(0);
      if (lane < 32) {                            // y column (col 256)
        xt[lane][256] = (_Float16)y[(size_t)t*BATCH + b0 + lane];
      }
    }
    __syncthreads();
    for (int u = wave; u < 32; u += 8) {
      const int m0 = (u >> 4) * 16, j0 = (u & 15) * 16;
      v8f aR = {}, aU = {}, aNi = {}, aNh = {};
      for (int kc = 0; kc < 9; ++kc) {            // K over 288 (padded 257)
        v16h a = load_a(&xt[m0][kc*32], KIB, lane);
        aR  = wmma_f16(a, load_b(Wib16 + (size_t)(      j0)*KIB + kc*32, KIB, lane), aR);
        aU  = wmma_f16(a, load_b(Wib16 + (size_t)(256 + j0)*KIB + kc*32, KIB, lane), aU);
        aNi = wmma_f16(a, load_b(Wib16 + (size_t)(512 + j0)*KIB + kc*32, KIB, lane), aNi);
      }
      for (int kc = 0; kc < 8; ++kc) {
        v16h a = load_a(&hbuf[cur][m0][kc*32], HD, lane);
        aR  = wmma_f16(a, load_b(Whb16 + (size_t)(      j0)*HD + kc*32, HD, lane), aR);
        aU  = wmma_f16(a, load_b(Whb16 + (size_t)(256 + j0)*HD + kc*32, HD, lane), aU);
        aNh = wmma_f16(a, load_b(Whb16 + (size_t)(512 + j0)*HD + kc*32, HD, lane), aNh);
      }
      const int j  = j0 + (lane & 15);
      const float bR  = bih[j] + bhh[j];
      const float bU  = bih[256+j] + bhh[256+j];
      const float bNi = bih[512+j], bNh = bhh[512+j];
      const int mb = m0 + ((lane >> 4) << 3);
      #pragma unroll
      for (int i = 0; i < 8; ++i) {
        const int r = mb + i;
        float rr = sigm(aR[i] + bR);
        float uu = sigm(aU[i] + bU);
        float nn = tanhf(aNi[i] + bNi + rr*(aNh[i] + bNh));
        float hnew = (1.f - uu)*nn + uu*(float)hbuf[cur][r][j];
        hbuf[cur^1][r][j] = (_Float16)hnew;
        Aall[((size_t)t*BATCH + b0 + r)*HD + j] = (_Float16)hnew;
      }
    }
    __syncthreads();
    cur ^= 1;
  }
}

// ---------------------------------------------------------------------------
// K3 scan helpers: M = 16 rows, N = 64, regimes packed into wave map (8 waves
// -> wave = k*4 + nblk).
// ---------------------------------------------------------------------------
__device__ inline void gemm_l1(const _Float16* Hbase, int ldH,
                               const _Float16* Zls,
                               const _Float16* Wp /*[2][64][320]*/,
                               const float* b /*[2][64]*/,
                               _Float16 (*outb)[16][64], int wave, int lane) {
  const int k = wave >> 2, n0 = (wave & 3) * 16;
  const _Float16* w = Wp + ((size_t)k*64 + n0)*320;
  v8f acc = {};
  for (int kc = 0; kc < 8; ++kc)
    acc = wmma_f16(load_a(Hbase + kc*32, ldH, lane), load_b(w + kc*32, 320, lane), acc);
  for (int kc = 0; kc < 2; ++kc)
    acc = wmma_f16(load_a(Zls + kc*32, 64, lane), load_b(w + 256 + kc*32, 320, lane), acc);
  const int n = n0 + (lane & 15);
  const float bb = b[k*64 + n];
  const int mb = (lane >> 4) << 3;
  #pragma unroll
  for (int i = 0; i < 8; ++i) {
    float v = acc[i] + bb;
    outb[k][mb + i][n] = (_Float16)(v > 0.f ? v : 0.f);
  }
}
__device__ inline void gemm_64(_Float16 (*in)[16][64],
                               const _Float16* Wp /*[2][64][64]*/, const float* b,
                               _Float16 (*outb)[16][64], int wave, int lane) {
  const int k = wave >> 2, n0 = (wave & 3) * 16;
  const _Float16* w = Wp + ((size_t)k*64 + n0)*64;
  v8f acc = {};
  for (int kc = 0; kc < 2; ++kc)
    acc = wmma_f16(load_a(&in[k][0][kc*32], 64, lane), load_b(w + kc*32, 64, lane), acc);
  const int n = n0 + (lane & 15);
  const float bb = b[k*64 + n];
  const int mb = (lane >> 4) << 3;
  #pragma unroll
  for (int i = 0; i < 8; ++i) {
    float v = acc[i] + bb;
    outb[k][mb + i][n] = (_Float16)(v > 0.f ? v : 0.f);
  }
}
__device__ inline void gemm_heads(_Float16 (*in)[16][64],
                                  const _Float16* Wm, const float* bm,
                                  const _Float16* Ws, const float* bs,
                                  float (*om)[16][64], float (*os)[16][64],
                                  int wave, int lane) {
  const int k = wave >> 2, n0 = (wave & 3) * 16;
  const _Float16* wm = Wm + ((size_t)k*64 + n0)*64;
  const _Float16* ws = Ws + ((size_t)k*64 + n0)*64;
  v8f am = {}, as = {};
  for (int kc = 0; kc < 2; ++kc) {
    v16h a = load_a(&in[k][0][kc*32], 64, lane);
    am = wmma_f16(a, load_b(wm + kc*32, 64, lane), am);
    as = wmma_f16(a, load_b(ws + kc*32, 64, lane), as);
  }
  const int n = n0 + (lane & 15);
  const float bmv = bm[k*64 + n], bsv = bs[k*64 + n];
  const int mb = (lane >> 4) << 3;
  #pragma unroll
  for (int i = 0; i < 8; ++i) {
    om[k][mb + i][n] = am[i] + bmv;
    os[k][mb + i][n] = softplusf(as[i] + bsv);
  }
}

// ---------------------------------------------------------------------------
// K3: sequential scan. 16 batch rows / WG, 32 WGs.
// ---------------------------------------------------------------------------
__global__ __launch_bounds__(256) void scan_kernel(
    const float* __restrict__ y, const float* __restrict__ Wdpost,
    const float* __restrict__ bdpost,
    const float* __restrict__ bzt1, const float* __restrict__ bzt2,
    const float* __restrict__ bztm, const float* __restrict__ bzts,
    const float* __restrict__ bzq1, const float* __restrict__ bzq2,
    const float* __restrict__ bzqm, const float* __restrict__ bzqs,
    const float* __restrict__ Wye1, const float* __restrict__ bye1,
    const float* __restrict__ Wye2, const float* __restrict__ bye2,
    const float* __restrict__ Wyem, const float* __restrict__ byem,
    const float* __restrict__ Wyes, const float* __restrict__ byes,
    const float* __restrict__ eps, const float* __restrict__ d0_oh,
    const unsigned* __restrict__ keys,
    const _Float16* __restrict__ hall, const _Float16* __restrict__ Aall,
    const _Float16* __restrict__ Wzt1p, const _Float16* __restrict__ Wzt2p,
    const _Float16* __restrict__ Wztmp, const _Float16* __restrict__ Wztsp,
    const _Float16* __restrict__ Wzq1p, const _Float16* __restrict__ Wzq2p,
    const _Float16* __restrict__ Wzqmp, const _Float16* __restrict__ Wzqsp,
    const float* __restrict__ TransWs, float* __restrict__ partials,
    float* __restrict__ o_ym, float* __restrict__ o_ys,
    float* __restrict__ o_zs, float* __restrict__ o_dp) {
  __shared__ float    zprev[16][64];
  __shared__ _Float16 z16[16][64];
  __shared__ _Float16 hid[2][16][64];
  __shared__ _Float16 zp[2][16][64];
  __shared__ float pm[2][16][64], ps[2][16][64], qm[2][16][64], qs[2][16][64];
  __shared__ float dpk[2][16][2], dpost[16][2], dpp[16][2];
  __shared__ int   didx[16];
  __shared__ float em[2][16], es[2][16];
  __shared__ float Tr[2][2];
  __shared__ float red[256];
  const int b0 = blockIdx.x * 16;
  const int tid = threadIdx.x, lane = tid & 31, wave = tid >> 5;
  if (tid < 16) {
    didx[tid] = (d0_oh[(size_t)(b0 + tid)*2 + 1] > 0.5f) ? 1 : 0;
    dpp[tid][0] = 0.5f; dpp[tid][1] = 0.5f;
  }
  if (tid < 4) ((float*)Tr)[tid] = TransWs[tid];
  for (int i = tid; i < 16*64; i += 256) {
    ((float*)zprev)[i] = 0.f;
    ((_Float16*)z16)[i] = (_Float16)0.f;
  }
  float accKG = 0.f, accKC = 0.f, accNL = 0.f;
  __syncthreads();
  for (int t = 0; t < T_LEN; ++t) {
    const _Float16* Ht = hall + ((size_t)t*BATCH + b0)*HD;
    const _Float16* At = Aall + ((size_t)t*BATCH + b0)*HD;
    if (tid == 0 && t + 1 < T_LEN) {
      __builtin_prefetch(Aall + ((size_t)(t+1)*BATCH + b0)*HD, 0, 1);
      __builtin_prefetch(hall + ((size_t)(t+1)*BATCH + b0)*HD, 0, 1);
    }
    // regime posterior logits: A_t @ Wdpost + bdpost
    if (tid < 64) {
      const int k = tid >> 5, b = (tid >> 1) & 15, d = tid & 1;
      const _Float16* a = At + (size_t)b*HD;
      float s = 0.f;
      for (int h = 0; h < HD; ++h)
        s += (float)a[h] * Wdpost[((size_t)k*HD + h)*2 + d];
      dpk[k][b][d] = s + bdpost[k*2 + d];
    }
    __syncthreads();
    if (tid < 32) {                            // softmax over d per (k,b)
      const int k = tid >> 4, b = tid & 15;
      float l0 = dpk[k][b][0], l1 = dpk[k][b][1];
      float m = fmaxf(l0, l1);
      float e0 = expf(l0 - m), e1 = expf(l1 - m), s = e0 + e1;
      dpk[k][b][0] = e0/s; dpk[k][b][1] = e1/s;
    }
    __syncthreads();
    if (tid < 16) {                            // mix + categorical sample (JAX-exact)
      const int b = tid, kp = didx[b];
      float p0 = dpk[kp][b][0], p1 = dpk[kp][b][1];
      dpost[b][0] = p0; dpost[b][1] = p1;
      o_dp[((size_t)t*BATCH + b0 + b)*2 + 0] = p0;
      o_dp[((size_t)t*BATCH + b0 + b)*2 + 1] = p1;
      const unsigned kk0 = keys[(size_t)t*2], kk1 = keys[(size_t)t*2 + 1];
      float g[2];
      #pragma unroll
      for (int d = 0; d < 2; ++d) {
        unsigned j = 2u*(unsigned)(b0 + b) + (unsigned)d;
        unsigned x0, x1, r0, r1;
        if (j < 512u) { x0 = j; x1 = j + 512u; } else { x0 = j - 512u; x1 = j; }
        threefry2x32(kk0, kk1, x0, x1, r0, r1);
        unsigned bits = (j < 512u) ? r0 : r1;
        float f = __uint_as_float((bits >> 9) | 0x3f800000u) - 1.0f;
        float u = fmaxf(1.17549435e-38f, f + 1.17549435e-38f);
        g[d] = -logf(-logf(u));
      }
      float s0 = logf(p0) + g[0], s1 = logf(p1) + g[1];
      didx[b] = (s1 > s0) ? 1 : 0;
    } else if (tid < 64) {                     // categorical KL (uses prev dpp)
      const int k = (tid - 32) >> 4, b = (tid - 32) & 15;
      float s = 0.f;
      #pragma unroll
      for (int d = 0; d < 2; ++d) {
        float p = dpk[k][b][d];
        s += p * logf(p / Tr[k][d]);
      }
      accKC += dpp[b][k] * s;
    }
    __syncthreads();
    if (tid < 32) { const int b = tid >> 1, d = tid & 1; dpp[b][d] = dpost[b][d]; }
    // prior branch: [h_t | z_prev]
    gemm_l1(Ht, HD, &z16[0][0], Wzt1p, bzt1, hid, wave, lane);
    __syncthreads();
    gemm_64(hid, Wzt2p, bzt2, zp, wave, lane);
    __syncthreads();
    gemm_heads(zp, Wztmp, bztm, Wztsp, bzts, pm, ps, wave, lane);
    __syncthreads();
    // posterior branch: [A_t | z_prev]
    gemm_l1(At, HD, &z16[0][0], Wzq1p, bzq1, hid, wave, lane);
    __syncthreads();
    gemm_64(hid, Wzq2p, bzq2, zp, wave, lane);
    __syncthreads();
    gemm_heads(zp, Wzqmp, bzqm, Wzqsp, bzqs, qm, qs, wave, lane);
    __syncthreads();
    // reparameterized z_t
    for (int i = tid; i < 16*64; i += 256) {
      const int b = i >> 6, z = i & 63, kn = didx[b];
      float zt = qm[kn][b][z] +
                 eps[((size_t)t*BATCH + b0 + b)*ZD + z] * qs[kn][b][z];
      zprev[b][z] = zt;
      z16[b][z] = (_Float16)zt;
      o_zs[((size_t)t*BATCH + b0 + b)*ZD + z] = zt;
    }
    // gaussian KL, weighted by d_post
    for (int i = tid; i < 2*16*64; i += 256) {
      const int k = i >> 10, b = (i >> 6) & 15, z = i & 63;
      float P = ps[k][b][z], Q = qs[k][b][z], dm = qm[k][b][z] - pm[k][b][z];
      accKG += dpost[b][k] * 0.5f *
               (2.f*logf(P) - 2.f*logf(Q) + (Q*Q + dm*dm)/(P*P) - 1.f);
    }
    __syncthreads();
    // emission head (hidden width = 1): [h_t | z_t]
    if (tid < 32) {
      const int k = tid >> 4, b = tid & 15;
      const _Float16* h_ = Ht + (size_t)b*HD;
      float s = 0.f;
      for (int i = 0; i < HD; ++i) s += (float)h_[i] * Wye1[(size_t)k*320 + i];
      for (int z = 0; z < ZD; ++z) s += zprev[b][z] * Wye1[(size_t)k*320 + 256 + z];
      float hdn = fmaxf(s + bye1[k], 0.f);
      float z2  = fmaxf(hdn * Wye2[k] + bye2[k], 0.f);
      em[k][b] = z2 * Wyem[k] + byem[k];
      es[k][b] = softplusf(z2 * Wyes[k] + byes[k]);
    }
    __syncthreads();
    if (tid < 16) {
      const int b = tid, kn = didx[b];
      o_ym[(size_t)t*BATCH + b0 + b] = em[kn][b];
      o_ys[(size_t)t*BATCH + b0 + b] = es[kn][b];
      const float yt = y[(size_t)t*BATCH + b0 + b];
      #pragma unroll
      for (int k = 0; k < 2; ++k) {
        float E = em[k][b], S = es[k][b];
        accNL += dpost[b][k] *
                 (0.91893853320467274f + logf(S) + (yt - E)*(yt - E)/(2.f*S*S));
      }
    }
    __syncthreads();
  }
  // per-WG loss reduction (deterministic: summed by k_final)
  red[tid] = accKG; __syncthreads();
  for (int s = 128; s > 0; s >>= 1) { if (tid < s) red[tid] += red[tid + s]; __syncthreads(); }
  if (tid == 0) partials[blockIdx.x*3 + 0] = red[0];
  __syncthreads();
  red[tid] = accKC; __syncthreads();
  for (int s = 128; s > 0; s >>= 1) { if (tid < s) red[tid] += red[tid + s]; __syncthreads(); }
  if (tid == 0) partials[blockIdx.x*3 + 1] = red[0];
  __syncthreads();
  red[tid] = accNL; __syncthreads();
  for (int s = 128; s > 0; s >>= 1) { if (tid < s) red[tid] += red[tid + s]; __syncthreads(); }
  if (tid == 0) partials[blockIdx.x*3 + 2] = red[0];
}

__global__ void k_final(const float* __restrict__ partials, float* __restrict__ out) {
  const int i = threadIdx.x;
  if (i < 3) {
    float s = 0.f;
    for (int w = 0; w < 32; ++w) s += partials[w*3 + i];
    out[i] = s;
  }
}

// ---------------------------------------------------------------------------
extern "C" void kernel_launch(void* const* d_in, const int* in_sizes, int n_in,
                              void* d_out, int out_size, void* d_ws, size_t ws_size,
                              hipStream_t stream) {
  (void)in_sizes; (void)n_in; (void)out_size; (void)ws_size;
  const float* x       = (const float*)d_in[0];
  const float* y       = (const float*)d_in[1];
  const float* Wdprior = (const float*)d_in[2];
  const float* bdprior = (const float*)d_in[3];
  const float* Wdpost  = (const float*)d_in[4];
  const float* bdpost  = (const float*)d_in[5];
  const float* Wzt1 = (const float*)d_in[6];  const float* bzt1 = (const float*)d_in[7];
  const float* Wzt2 = (const float*)d_in[8];  const float* bzt2 = (const float*)d_in[9];
  const float* Wztm = (const float*)d_in[10]; const float* bztm = (const float*)d_in[11];
  const float* Wzts = (const float*)d_in[12]; const float* bzts = (const float*)d_in[13];
  const float* Wzq1 = (const float*)d_in[14]; const float* bzq1 = (const float*)d_in[15];
  const float* Wzq2 = (const float*)d_in[16]; const float* bzq2 = (const float*)d_in[17];
  const float* Wzqm = (const float*)d_in[18]; const float* bzqm = (const float*)d_in[19];
  const float* Wzqs = (const float*)d_in[20]; const float* bzqs = (const float*)d_in[21];
  const float* Wye1 = (const float*)d_in[22]; const float* bye1 = (const float*)d_in[23];
  const float* Wye2 = (const float*)d_in[24]; const float* bye2 = (const float*)d_in[25];
  const float* Wyem = (const float*)d_in[26]; const float* byem = (const float*)d_in[27];
  const float* Wyes = (const float*)d_in[28]; const float* byes = (const float*)d_in[29];
  const float* Wih_f = (const float*)d_in[30]; const float* Whh_f = (const float*)d_in[31];
  const float* bih_f = (const float*)d_in[32]; const float* bhh_f = (const float*)d_in[33];
  const float* Wih_b = (const float*)d_in[34]; const float* Whh_b = (const float*)d_in[35];
  const float* bih_b = (const float*)d_in[36]; const float* bhh_b = (const float*)d_in[37];
  const float* eps   = (const float*)d_in[38];
  const float* d0_oh = (const float*)d_in[39];
  const unsigned* keys = (const unsigned*)d_in[40];
  float* out = (float*)d_out;
  float* o_ym = out + 3;
  float* o_ys = o_ym + (size_t)T_LEN*BATCH;
  float* o_zs = o_ys + (size_t)T_LEN*BATCH;
  float* o_dp = o_zs + (size_t)T_LEN*BATCH*ZD;

  // workspace carve (256B aligned)
  char* wsb = (char*)d_ws;
  size_t off = 0;
  auto carve = [&](size_t bytes) -> char* {
    char* p = wsb + off;
    off = (off + bytes + 255) & ~(size_t)255;
    return p;
  };
  float*    TransWs  = (float*)carve(4*sizeof(float));
  float*    partials = (float*)carve(32*3*sizeof(float));
  _Float16* hall  = (_Float16*)carve((size_t)T_LEN*BATCH*HD*2);
  _Float16* Aall  = (_Float16*)carve((size_t)T_LEN*BATCH*HD*2);
  _Float16* x16   = (_Float16*)carve((size_t)T_LEN*BATCH*XD*2);
  _Float16* Wih16 = (_Float16*)carve((size_t)G3*XD*2);
  _Float16* Whh16 = (_Float16*)carve((size_t)G3*HD*2);
  _Float16* Wib16 = (_Float16*)carve((size_t)G3*KIB*2);
  _Float16* Whb16 = (_Float16*)carve((size_t)G3*HD*2);
  _Float16* Wzt1p = (_Float16*)carve((size_t)2*320*64*2);
  _Float16* Wzq1p = (_Float16*)carve((size_t)2*320*64*2);
  _Float16* Wzt2p = (_Float16*)carve((size_t)2*64*64*2);
  _Float16* Wztmp = (_Float16*)carve((size_t)2*64*64*2);
  _Float16* Wztsp = (_Float16*)carve((size_t)2*64*64*2);
  _Float16* Wzq2p = (_Float16*)carve((size_t)2*64*64*2);
  _Float16* Wzqmp = (_Float16*)carve((size_t)2*64*64*2);
  _Float16* Wzqsp = (_Float16*)carve((size_t)2*64*64*2);

  // K0: prep
  hipLaunchKernelGGL(k_trans, dim3(1), dim3(64), 0, stream, Wdprior, bdprior, TransWs, out);
  hipLaunchKernelGGL(k_cvt,     dim3(2048), dim3(256), 0, stream, x, x16, T_LEN*BATCH*XD);
  hipLaunchKernelGGL(k_cvt,     dim3(96),  dim3(256), 0, stream, Wih_f, Wih16, G3*XD);
  hipLaunchKernelGGL(k_cvt,     dim3(192), dim3(256), 0, stream, Whh_f, Whh16, G3*HD);
  hipLaunchKernelGGL(k_cvt,     dim3(192), dim3(256), 0, stream, Whh_b, Whb16, G3*HD);
  hipLaunchKernelGGL(k_packwib, dim3(192), dim3(256), 0, stream, Wih_b, Wib16);
  hipLaunchKernelGGL(k_packT,   dim3(80),  dim3(256), 0, stream, Wzt1, Wzt1p, 320, 64);
  hipLaunchKernelGGL(k_packT,   dim3(80),  dim3(256), 0, stream, Wzq1, Wzq1p, 320, 64);
  hipLaunchKernelGGL(k_packT,   dim3(32),  dim3(256), 0, stream, Wzt2, Wzt2p, 64, 64);
  hipLaunchKernelGGL(k_packT,   dim3(32),  dim3(256), 0, stream, Wztm, Wztmp, 64, 64);
  hipLaunchKernelGGL(k_packT,   dim3(32),  dim3(256), 0, stream, Wzts, Wztsp, 64, 64);
  hipLaunchKernelGGL(k_packT,   dim3(32),  dim3(256), 0, stream, Wzq2, Wzq2p, 64, 64);
  hipLaunchKernelGGL(k_packT,   dim3(32),  dim3(256), 0, stream, Wzqm, Wzqmp, 64, 64);
  hipLaunchKernelGGL(k_packT,   dim3(32),  dim3(256), 0, stream, Wzqs, Wzqsp, 64, 64);
  // K1/K2: GRU recurrences (batch-parallel across WGs, sequential in t)
  hipLaunchKernelGGL(gru_fwd, dim3(16), dim3(256), 0, stream,
                     x16, bih_f, bhh_f, Wih16, Whh16, hall);
  hipLaunchKernelGGL(gru_bwd, dim3(16), dim3(256), 0, stream,
                     y, bih_b, bhh_b, Wib16, Whb16, hall, Aall);
  // K3: scan
  hipLaunchKernelGGL(scan_kernel, dim3(32), dim3(256), 0, stream,
                     y, Wdpost, bdpost,
                     bzt1, bzt2, bztm, bzts,
                     bzq1, bzq2, bzqm, bzqs,
                     Wye1, bye1, Wye2, bye2, Wyem, byem, Wyes, byes,
                     eps, d0_oh, keys, hall, Aall,
                     Wzt1p, Wzt2p, Wztmp, Wztsp,
                     Wzq1p, Wzq2p, Wzqmp, Wzqsp,
                     TransWs, partials, o_ym, o_ys, o_zs, o_dp);
  // K4: deterministic loss finalize
  hipLaunchKernelGGL(k_final, dim3(1), dim3(32), 0, stream, partials, out);
}
